// RNAAttention_65816078844220
// MI455X (gfx1250) — matrix operand order repaired and checked by reading, not compiled, not run on hardware
//
#include <hip/hip_runtime.h>
#include <hip/hip_bf16.h>

#define B_   4
#define S_   2048
#define HID_ 1024
#define NH_  16
#define HD_  64

typedef __attribute__((ext_vector_type(16))) __bf16 v16bf;
typedef __attribute__((ext_vector_type(8)))  __bf16 v8bf;
typedef __attribute__((ext_vector_type(4)))  __bf16 v4bf;
typedef __attribute__((ext_vector_type(8)))  float  v8f;
typedef __attribute__((ext_vector_type(4)))  float  v4f;

// ---- helpers ---------------------------------------------------------------

static __device__ __forceinline__ v8bf ld8(const __bf16* p) {
    return *reinterpret_cast<const v8bf*>(p);
}

static __device__ __forceinline__ v16bf cat8(v8bf lo, v8bf hi) {
    v16bf r;
#pragma unroll
    for (int i = 0; i < 8; ++i) { r[i] = lo[i]; r[i + 8] = hi[i]; }
    return r;
}

static __device__ __forceinline__ v8f wmma_bf16(v16bf a, v16bf b, v8f c) {
    // D = A(16x32 bf16) x B(32x16 bf16) + C(16x16 f32)
    return __builtin_amdgcn_wmma_f32_16x16x32_bf16(
        false, a, false, b, (short)0, c, false, false);
}

// ---- fp32 -> bf16 conversion ----------------------------------------------

__global__ void cvt_bf16_kernel(const float* __restrict__ in,
                                __bf16* __restrict__ out, int n4) {
    int i = blockIdx.x * blockDim.x + threadIdx.x;
    if (i < n4) {
        v4f x = reinterpret_cast<const v4f*>(in)[i];
        v4bf y;
        y[0] = (__bf16)x[0]; y[1] = (__bf16)x[1];
        y[2] = (__bf16)x[2]; y[3] = (__bf16)x[3];
        reinterpret_cast<v4bf*>(out)[i] = y;
    }
}

// ---- QKV projection + RoPE -------------------------------------------------
// grid: (M/64=128, HID/64=16, 3)   block: 128 (4 waves), wave tile = 16x64
// z=0: Q (RoPE, -> [B,NH,S,D] bf16)  z=1: K (RoPE, same)  z=2: V (-> [B,NH,D,S])

__global__ __launch_bounds__(128) void qkv_rope_kernel(
    const __bf16* __restrict__ X,   // [B*S, HID] bf16
    const __bf16* __restrict__ Wq,  // [HID, HID] bf16 (row n holds W[n][k])
    const __bf16* __restrict__ Wk,
    const __bf16* __restrict__ Wv,
    const int*    __restrict__ pos, // [B*S]
    __bf16* __restrict__ Qo, __bf16* __restrict__ Ko, __bf16* __restrict__ VTo)
{
    const int lane = threadIdx.x & 31, wave = threadIdx.x >> 5;
    const int half = lane >> 4, nn = lane & 15;
    const int m0   = blockIdx.x * 64 + wave * 16;
    const int col0 = blockIdx.y * 64;                 // head-aligned
    const int z    = blockIdx.z;
    const __bf16* W = (z == 0) ? Wq : ((z == 1) ? Wk : Wv);
    const int base1 = half * 16;   // A-fragment K offset
    const int baseB = half * 16;   // B-fragment K offset
    const int mrow  = m0 + nn;     // A row owned by this lane

    v8f acc[4] = {v8f{}, v8f{}, v8f{}, v8f{}};
    const __bf16* xr = X + (size_t)mrow * HID_;
    const __bf16* w0 = W + (size_t)(col0 + nn) * HID_;

    for (int k0 = 0; k0 < HID_; k0 += 32) {
        v16bf a = cat8(ld8(xr + k0 + base1), ld8(xr + k0 + base1 + 8));
#pragma unroll
        for (int n = 0; n < 4; ++n) {
            const __bf16* wp = w0 + (size_t)n * 16 * HID_ + k0 + baseB;
            v16bf bm = cat8(ld8(wp), ld8(wp + 8));
            acc[n] = wmma_bf16(a, bm, acc[n]);
        }
    }

    const int h = blockIdx.y;   // col0/64: this wave's head
    if (z < 2) {
        __bf16* O = (z == 0) ? Qo : Ko;
#pragma unroll
        for (int j = 0; j < 8; ++j) {
            int m = m0 + half * 8 + j;
            int b = m >> 11;             // /S_
            int s = m & (S_ - 1);
            float p = (float)pos[m];
#pragma unroll
            for (int n = 0; n < 2; ++n) {
                int dlo = n * 16 + nn;   // 0..31
                float inv = __powf(10000.0f, -(float)dlo * (1.0f / 32.0f));
                float si, co;
                __sincosf(p * inv, &si, &co);
                float xlo = acc[n][j], xhi = acc[n + 2][j];
                float rlo = xlo * co - xhi * si;   // x*cos + rot_half(x)*sin
                float rhi = xhi * co + xlo * si;
                size_t ob = ((size_t)(b * NH_ + h) * S_ + s) * HD_;
                O[ob + dlo]      = (__bf16)rlo;
                O[ob + dlo + 32] = (__bf16)rhi;
            }
        }
    } else {
#pragma unroll
        for (int n = 0; n < 4; ++n) {
            int d = n * 16 + nn;
#pragma unroll
            for (int j = 0; j < 8; ++j) {
                int m = m0 + half * 8 + j;
                int b = m >> 11;
                int s = m & (S_ - 1);
                VTo[((size_t)(b * NH_ + h) * HD_ + d) * S_ + s] = (__bf16)acc[n][j];
            }
        }
    }
}

// ---- flash attention -------------------------------------------------------
// grid: (B*NH=64, S/64=32)  block: 128 (4 waves); each wave owns a 16-row q-block.

__global__ __launch_bounds__(128) void attn_kernel(
    const __bf16* __restrict__ Q,    // [B,NH,S,D] bf16
    const __bf16* __restrict__ K,    // [B,NH,S,D] bf16
    const __bf16* __restrict__ VT,   // [B,NH,D,S] bf16
    const float*  __restrict__ mask, // [B,1,S,S] f32
    __bf16* __restrict__ AO)         // [B,S,HID] bf16
{
    __shared__ __bf16 plds[4][16][32] __attribute__((aligned(64)));

    const int lane = threadIdx.x & 31, wave = threadIdx.x >> 5;
    const int half = lane >> 4, nn = lane & 15;
    const int bh = blockIdx.x, b = bh >> 4, h = bh & 15;
    const int q0 = (blockIdx.y * 4 + wave) * 16;
    const int base1 = half * 16, baseB = half * 16;

    const __bf16* Qb = Q  + (size_t)bh * S_ * HD_;
    const __bf16* Kb = K  + (size_t)bh * S_ * HD_;
    const __bf16* Vb = VT + (size_t)bh * HD_ * S_;
    const float*  Mb = mask + (size_t)b * S_ * S_;

    // Q fragments held for the whole pass (contraction over d)
    const __bf16* qp = Qb + (size_t)(q0 + nn) * HD_;
    v16bf aq0 = cat8(ld8(qp + base1),      ld8(qp + base1 + 8));
    v16bf aq1 = cat8(ld8(qp + 32 + base1), ld8(qp + 32 + base1 + 8));

    v8f o[4] = {v8f{}, v8f{}, v8f{}, v8f{}};
    float mrun[8], lrun[8];
#pragma unroll
    for (int j = 0; j < 8; ++j) { mrun[j] = -3.0e38f; lrun[j] = 0.0f; }

    for (int kk0 = 0; kk0 < S_; kk0 += 32) {
        // ---- scores: two 16x16 tiles over columns kk0..kk0+31
        v8f s0 = {}, s1 = {};
        {
            const __bf16* kp0 = Kb + (size_t)(kk0 + nn) * HD_;
            const __bf16* kp1 = Kb + (size_t)(kk0 + 16 + nn) * HD_;
            v16bf b00 = cat8(ld8(kp0 + baseB),      ld8(kp0 + baseB + 8));
            v16bf b01 = cat8(ld8(kp0 + 32 + baseB), ld8(kp0 + 32 + baseB + 8));
            v16bf b10 = cat8(ld8(kp1 + baseB),      ld8(kp1 + baseB + 8));
            v16bf b11 = cat8(ld8(kp1 + 32 + baseB), ld8(kp1 + 32 + baseB + 8));
            s0 = wmma_bf16(aq0, b00, s0);
            s0 = wmma_bf16(aq1, b01, s0);
            s1 = wmma_bf16(aq0, b10, s1);
            s1 = wmma_bf16(aq1, b11, s1);
        }

        // ---- online softmax (per-row stats; rows j+8*half per lane-half)
        float p0[8], p1[8], f[8];
#pragma unroll
        for (int j = 0; j < 8; ++j) {
            int q = q0 + half * 8 + j;
            float v0 = s0[j] * 0.125f + Mb[(size_t)q * S_ + kk0 + nn];
            float v1 = s1[j] * 0.125f + Mb[(size_t)q * S_ + kk0 + 16 + nn];
            float mx = fmaxf(v0, v1);
#pragma unroll
            for (int d = 1; d < 16; d <<= 1) mx = fmaxf(mx, __shfl_xor(mx, d, 32));
            float mnew = fmaxf(mrun[j], mx);
            f[j] = __expf(mrun[j] - mnew);
            float e0 = __expf(v0 - mnew), e1 = __expf(v1 - mnew);
            float rs = e0 + e1;
#pragma unroll
            for (int d = 1; d < 16; d <<= 1) rs += __shfl_xor(rs, d, 32);
            lrun[j] = lrun[j] * f[j] + rs;
            mrun[j] = mnew;
            p0[j] = e0; p1[j] = e1;
        }

        // rescale accumulators + stash P (C-layout -> row-major bf16 in LDS)
#pragma unroll
        for (int j = 0; j < 8; ++j) {
            o[0][j] *= f[j]; o[1][j] *= f[j]; o[2][j] *= f[j]; o[3][j] *= f[j];
            plds[wave][half * 8 + j][nn]      = (__bf16)p0[j];
            plds[wave][half * 8 + j][16 + nn] = (__bf16)p1[j];
        }
        asm volatile("s_wait_dscnt 0" ::: "memory");  // DS in-order per wave

        // ---- PV: A = P(16x32) from LDS, B = VT rows (contiguous in kk)
        v16bf ap = cat8(ld8(&plds[wave][nn][base1]),
                        ld8(&plds[wave][nn][base1 + 8]));
        const __bf16* vp = Vb + (size_t)nn * S_ + kk0 + baseB;
#pragma unroll
        for (int n = 0; n < 4; ++n) {
            const __bf16* vpn = vp + (size_t)n * 16 * S_;
            v16bf bv = cat8(ld8(vpn), ld8(vpn + 8));
            o[n] = wmma_bf16(ap, bv, o[n]);
        }
    }

    // ---- epilogue: normalize and store [B,S,NH*HD]
#pragma unroll
    for (int j = 0; j < 8; ++j) {
        int q = q0 + half * 8 + j;
        float inv = 1.0f / lrun[j];
        size_t ob = ((size_t)b * S_ + q) * HID_ + h * HD_;
#pragma unroll
        for (int n = 0; n < 4; ++n)
            AO[ob + n * 16 + nn] = (__bf16)(o[n][j] * inv);
    }
}

// ---- output projection -----------------------------------------------------
// grid: (128, 16)  block: 128; C[m][n] = sum_k A[m][k] * Wo[n][k], fp32 out.

__global__ __launch_bounds__(128) void oproj_kernel(
    const __bf16* __restrict__ A,   // [B*S, HID] bf16
    const __bf16* __restrict__ Wo,  // [HID, HID] bf16
    float* __restrict__ out)        // [B*S, HID] f32
{
    const int lane = threadIdx.x & 31, wave = threadIdx.x >> 5;
    const int half = lane >> 4, nn = lane & 15;
    const int m0   = blockIdx.x * 64 + wave * 16;
    const int col0 = blockIdx.y * 64;
    const int base1 = half * 16, baseB = half * 16;
    const int mrow  = m0 + nn;

    v8f acc[4] = {v8f{}, v8f{}, v8f{}, v8f{}};
    const __bf16* ar = A  + (size_t)mrow * HID_;
    const __bf16* w0 = Wo + (size_t)(col0 + nn) * HID_;

    for (int k0 = 0; k0 < HID_; k0 += 32) {
        v16bf a = cat8(ld8(ar + k0 + base1), ld8(ar + k0 + base1 + 8));
#pragma unroll
        for (int n = 0; n < 4; ++n) {
            const __bf16* wp = w0 + (size_t)n * 16 * HID_ + k0 + baseB;
            v16bf bm = cat8(ld8(wp), ld8(wp + 8));
            acc[n] = wmma_bf16(a, bm, acc[n]);
        }
    }
#pragma unroll
    for (int n = 0; n < 4; ++n) {
#pragma unroll
        for (int j = 0; j < 8; ++j) {
            int m = m0 + half * 8 + j;
            out[(size_t)m * HID_ + col0 + n * 16 + nn] = acc[n][j];
        }
    }
}

// ---- launcher --------------------------------------------------------------

extern "C" void kernel_launch(void* const* d_in, const int* in_sizes, int n_in,
                              void* d_out, int out_size, void* d_ws, size_t ws_size,
                              hipStream_t stream) {
    (void)in_sizes; (void)n_in; (void)out_size; (void)ws_size;

    const float* hs   = (const float*)d_in[0];
    const float* mask = (const float*)d_in[1];
    const int*   pos  = (const int*)d_in[2];
    const float* Wq   = (const float*)d_in[3];
    const float* Wk   = (const float*)d_in[4];
    const float* Wv   = (const float*)d_in[5];
    const float* Wo   = (const float*)d_in[6];
    float* out = (float*)d_out;

    char* ws = (char*)d_ws;
    size_t off = 0;
    const size_t act_bytes = (size_t)B_ * S_ * HID_ * sizeof(__bf16); // 16 MB
    const size_t w_bytes   = (size_t)HID_ * HID_ * sizeof(__bf16);    //  2 MB
    __bf16* x_bf    = (__bf16*)(ws + off); off += act_bytes;
    __bf16* wq_bf   = (__bf16*)(ws + off); off += w_bytes;
    __bf16* wk_bf   = (__bf16*)(ws + off); off += w_bytes;
    __bf16* wv_bf   = (__bf16*)(ws + off); off += w_bytes;
    __bf16* wo_bf   = (__bf16*)(ws + off); off += w_bytes;
    __bf16* q_bf    = (__bf16*)(ws + off); off += act_bytes;
    __bf16* k_bf    = (__bf16*)(ws + off); off += act_bytes;
    __bf16* vt_bf   = (__bf16*)(ws + off); off += act_bytes;
    __bf16* attn_bf = (__bf16*)(ws + off); off += act_bytes;

    int n4 = B_ * S_ * HID_ / 4;
    cvt_bf16_kernel<<<(n4 + 255) / 256, 256, 0, stream>>>(hs, x_bf, n4);
    n4 = HID_ * HID_ / 4;
    cvt_bf16_kernel<<<(n4 + 255) / 256, 256, 0, stream>>>(Wq, wq_bf, n4);
    cvt_bf16_kernel<<<(n4 + 255) / 256, 256, 0, stream>>>(Wk, wk_bf, n4);
    cvt_bf16_kernel<<<(n4 + 255) / 256, 256, 0, stream>>>(Wv, wv_bf, n4);
    cvt_bf16_kernel<<<(n4 + 255) / 256, 256, 0, stream>>>(Wo, wo_bf, n4);

    qkv_rope_kernel<<<dim3(B_ * S_ / 64, HID_ / 64, 3), 128, 0, stream>>>(
        x_bf, wq_bf, wk_bf, wv_bf, pos, q_bf, k_bf, vt_bf);

    attn_kernel<<<dim3(B_ * NH_, S_ / 64), 128, 0, stream>>>(
        q_bf, k_bf, vt_bf, mask, attn_bf);

    oproj_kernel<<<dim3(B_ * S_ / 64, HID_ / 64), 128, 0, stream>>>(
        attn_bf, wo_bf, out);
}
